// KGEmbedding_28810640621943
// MI455X (gfx1250) — matrix operand order repaired and checked by reading, not compiled, not run on hardware
//
#include <hip/hip_runtime.h>
#include <hip/hip_bf16.h>

typedef __attribute__((ext_vector_type(8)))  __bf16 v8bf;
typedef __attribute__((ext_vector_type(16))) __bf16 v16bf;
typedef __attribute__((ext_vector_type(8)))  float  v8f;

#define WMMA_BF16(a, b, c) \
  __builtin_amdgcn_wmma_f32_16x16x32_bf16(false, (a), false, (b), (short)0, (c), false, false)

__device__ __forceinline__ v16bf cat8(v8bf lo, v8bf hi) {
  return __builtin_shufflevector(lo, hi, 0,1,2,3,4,5,6,7,8,9,10,11,12,13,14,15);
}
// monotone float<->uint mapping (unsigned order == float order)
__device__ __forceinline__ unsigned f2key(float f) {
  unsigned b = __float_as_uint(f);
  return (b >> 31) ? ~b : (b | 0x80000000u);
}
__device__ __forceinline__ float key2f(unsigned k) {
  unsigned b = (k >> 31) ? (k ^ 0x80000000u) : ~k;
  return __uint_as_float(b);
}
// generic shared pointer -> 32-bit LDS byte offset (addrspace(3) truncation)
typedef __attribute__((address_space(3))) char* lds3_t;
__device__ __forceinline__ unsigned lds_off(const void* p) {
  return (unsigned)(unsigned long long)(lds3_t)(p);
}

// ---------------- f32 -> bf16 conversion ----------------
__global__ void k_cvt(const float* __restrict__ s, __bf16* __restrict__ d, int n) {
  int i = blockIdx.x * blockDim.x + threadIdx.x;
  if (i < n) d[i] = (__bf16)s[i];
}

// ---------------- V (16 heads x 8192 x 64) -> Vt (16 heads x 64 x 8192) ----------------
// LDS-tiled transpose; global->LDS leg uses gfx1250 async-to-LDS loads (ASYNCcnt).
__global__ __launch_bounds__(256) void k_vtrans(const __bf16* __restrict__ V,
                                                __bf16* __restrict__ Vt) {
  __shared__ __align__(16) __bf16 tile[64][72];   // 144B row pitch (16B multiple)
  const int h2 = blockIdx.x & 15;
  const int nt = blockIdx.x >> 4;                 // 0..127
  const int n0 = nt * 64;
  const int tid = threadIdx.x;

  // load 64 rows x 128B, one b128 per lane per iter, fully coalesced
#pragma unroll
  for (int it = 0; it < 2; ++it) {
    const int r = it * 32 + (tid >> 3);           // n within tile
    const int c = (tid & 7) * 8;                  // d (bf16 units)
    const __bf16* gp = V + ((size_t)(h2 * 8192 + n0 + r)) * 64 + c;
    const unsigned l = lds_off(&tile[r][c]);
    asm volatile("global_load_async_to_lds_b128 %0, %1, off"
                 :: "v"(l), "v"(gp) : "memory");
  }
  asm volatile("s_wait_asynccnt 0" ::: "memory"); // my copies done before I signal
  __syncthreads();                                 // => everyone's copies visible

  // store transposed: each thread writes 16 contiguous n for one d (coalesced)
  const int d  = tid >> 2;
  const int ns = (tid & 3) * 16;
  __bf16* op = Vt + ((size_t)(h2 * 64 + d)) * 8192 + n0 + ns;
#pragma unroll
  for (int k = 0; k < 16; ++k) op[k] = tile[ns + k][d];
}

// ---------------- Y = X @ W^T + bias (+resid); wave computes 64x64 (4x4 WMMA grid) ----------------
__global__ __launch_bounds__(256) void k_gemm_bf16(
    const __bf16* __restrict__ X, const __bf16* __restrict__ W,
    const float* __restrict__ bias, const float* __restrict__ resid,
    float* __restrict__ outF, __bf16* __restrict__ outB,
    int M, int N, int K)
{
  const int lane = threadIdx.x & 31, w = threadIdx.x >> 5;
  const int lr = lane & 15, hi = lane >> 4;
  const int ntn  = N >> 6;                        // 64-wide column tiles
  const int tile = blockIdx.x * 8 + w;
  const int tm = tile / ntn, tn = tile % ntn;

  const __bf16* arow[4];
  const __bf16* brow[4];
#pragma unroll
  for (int i = 0; i < 4; ++i)
    arow[i] = X + (size_t)(tm * 64 + i * 16 + lr) * K + 16 * hi;     // A: lane = M-row
#pragma unroll
  for (int j = 0; j < 4; ++j)
    brow[j] = W + (size_t)(tn * 64 + j * 16 + lr) * K + 16 * hi;     // B: lane = N-row

  v8f acc[4][4];
  v8f zero = {};
#pragma unroll
  for (int i = 0; i < 4; ++i)
#pragma unroll
    for (int j = 0; j < 4; ++j) acc[i][j] = zero;

  for (int k0 = 0; k0 < K; k0 += 32) {
    v16bf a[4], b[4];
#pragma unroll
    for (int i = 0; i < 4; ++i)
      a[i] = cat8(*(const v8bf*)(arow[i] + k0), *(const v8bf*)(arow[i] + k0 + 8));
#pragma unroll
    for (int j = 0; j < 4; ++j)
      b[j] = *(const v16bf*)(brow[j] + k0);
#pragma unroll
    for (int i = 0; i < 4; ++i)
#pragma unroll
      for (int j = 0; j < 4; ++j)
        acc[i][j] = WMMA_BF16(a[i], b[j], acc[i][j]);
  }

#pragma unroll
  for (int j = 0; j < 4; ++j) {
    const int n = tn * 64 + j * 16 + lr;
    const float bv = bias ? bias[n] : 0.f;
#pragma unroll
    for (int i = 0; i < 4; ++i) {
#pragma unroll
      for (int v = 0; v < 8; ++v) {
        const int m = tm * 64 + i * 16 + v + 8 * hi;   // C/D: M = vgpr + 8*(lane>>4)
        float val = acc[i][j][v] + bv;
        if (outB) {
          outB[(size_t)m * N + n] = (__bf16)val;
        } else {
          if (resid) val += resid[(size_t)m * N + n];
          outF[(size_t)m * N + n] = val;
        }
      }
    }
  }
}

// ---------------- fused QK-softmax-top4096-PV, one block per (h2, token-block) ----------------
__global__ __launch_bounds__(256) void k_attn(
    const __bf16* __restrict__ Qbf, const __bf16* __restrict__ Kbf,
    const __bf16* __restrict__ Vt,  const float* __restrict__ query,
    const int* __restrict__ mask,   __bf16* __restrict__ Hbf)
{
  const int h2 = blockIdx.x;            // Q-column / V head
  const int tb = blockIdx.y;            // token block (16 tokens)
  const int t0 = tb * 16;
  const int h  = tb >> 2;               // K head = t/64
  const int tid = threadIdx.x;
  const int lane = tid & 31, w = tid >> 5;
  const int lr = lane & 15, hi = lane >> 4;

  __shared__ unsigned hist[16][256];
  __shared__ unsigned rowMaxKey[16];
  __shared__ float rowSum[16], rowInv[16];
  __shared__ unsigned pref[16], rankR[16];
  __shared__ __align__(32) __bf16 stage[8][16 * 32];
  __shared__ float outTile[16 * 64];

  // Q A-fragments (K-dim = 64 -> two 16x16x32 fragments), reused by every pass
  const __bf16* qrow = Qbf + (size_t)(t0 + lr) * 1024 + h2 * 64;
  v16bf aq0 = cat8(*(const v8bf*)(qrow + 16 * hi),      *(const v8bf*)(qrow + 16 * hi + 8));
  v16bf aq1 = cat8(*(const v8bf*)(qrow + 32 + 16 * hi), *(const v8bf*)(qrow + 32 + 16 * hi + 8));

  const __bf16* Kh = Kbf + (size_t)h * 8192 * 64;
  const int nbase = w * 1024;           // each wave owns 1024 of the 8192 n's

  if (tid < 16) { rowMaxKey[tid] = 0u; rowSum[tid] = 0.f; pref[tid] = 0u; rankR[tid] = 4096u; }
  for (int i = tid; i < 16 * 256; i += 256) ((unsigned*)hist)[i] = 0u;
  __syncthreads();

  // ---- 4-level radix select (8 bits/level) of the 4096th-largest logit per row;
  //      QK recomputed by WMMA each level (L2-resident operands, flops ~free) ----
  for (int L = 0; L < 4; ++L) {
    unsigned pr8[8]; float rm8[8];
#pragma unroll
    for (int v = 0; v < 8; ++v) {
      const int t = v + 8 * hi;
      pr8[v] = pref[t];
      rm8[v] = key2f(rowMaxKey[t]);     // meaningful from L>=1
    }
    unsigned pmk[8]; float ps[8];
#pragma unroll
    for (int v = 0; v < 8; ++v) { pmk[v] = 0u; ps[v] = 0.f; }

    for (int it = 0; it < 64; ++it) {
      const int n0 = nbase + it * 16;
      const __bf16* bp = Kh + (size_t)(n0 + lr) * 64 + 16 * hi;
      v16bf b0 = *(const v16bf*)(bp);
      v16bf b1 = *(const v16bf*)(bp + 32);
      v8f acc = {};
      acc = WMMA_BF16(aq0, b0, acc);
      acc = WMMA_BF16(aq1, b1, acc);
#pragma unroll
      for (int v = 0; v < 8; ++v) {
        const int t = v + 8 * hi;
        const float l = acc[v] * 0.125f;          // 1/sqrt(64)
        const unsigned key = f2key(l);
        if (L == 0) {
          pmk[v] = key > pmk[v] ? key : pmk[v];
          atomicAdd(&hist[t][key >> 24], 1u);
        } else {
          if (L == 1) ps[v] += __expf(l - rm8[v]);
          if ((key >> (32 - 8 * L)) == pr8[v])
            atomicAdd(&hist[t][(key >> (24 - 8 * L)) & 255u], 1u);
        }
      }
    }
#pragma unroll
    for (int v = 0; v < 8; ++v) {
      const int t = v + 8 * hi;
      if (L == 0) atomicMax(&rowMaxKey[t], pmk[v]);
      if (L == 1) atomicAdd(&rowSum[t], ps[v]);
    }
    __syncthreads();
    if (tid < 16) {
      const int t = tid;
      unsigned R = rankR[t], cum = 0, sel = 0;
      for (int i = 255; i >= 0; --i) {
        unsigned c = hist[t][i];
        if (R <= cum + c) { sel = (unsigned)i; break; }
        cum += c;
      }
      rankR[t] = R - cum;
      pref[t]  = (pref[t] << 8) | sel;
      if (L == 1) rowInv[t] = 1.f / rowSum[t];
    }
    for (int i = tid; i < 16 * 256; i += 256) ((unsigned*)hist)[i] = 0u;
    __syncthreads();
  }

  for (int i = tid; i < 16 * 64; i += 256) outTile[i] = 0.f;
  __syncthreads();

  unsigned thr8[8]; float rmx8[8], inv8[8];
#pragma unroll
  for (int v = 0; v < 8; ++v) {
    const int t = v + 8 * hi;
    thr8[v] = pref[t];                   // exact 4096th-largest key
    rmx8[v] = key2f(rowMaxKey[t]);
    inv8[v] = rowInv[t];
  }

  // ---- P @ V: recompute logits, p = exp(l-max)/sum if key>=thr else 0; WMMA into 16x64 ----
  v8f zero = {};
  v8f accAV[4];
#pragma unroll
  for (int j = 0; j < 4; ++j) accAV[j] = zero;
  __bf16* st = &stage[w][0];

  for (int s = 0; s < 32; ++s) {
    const int n0 = nbase + s * 32;
#pragma unroll
    for (int q = 0; q < 2; ++q) {
      const int nq = n0 + q * 16;
      const __bf16* bp = Kh + (size_t)(nq + lr) * 64 + 16 * hi;
      v16bf b0 = *(const v16bf*)(bp);
      v16bf b1 = *(const v16bf*)(bp + 32);
      v8f acc = {};
      acc = WMMA_BF16(aq0, b0, acc);
      acc = WMMA_BF16(aq1, b1, acc);
#pragma unroll
      for (int v = 0; v < 8; ++v) {
        const int t = v + 8 * hi;
        const float l = acc[v] * 0.125f;
        const unsigned key = f2key(l);
        const float p = (key >= thr8[v]) ? __expf(l - rmx8[v]) * inv8[v] : 0.f;
        st[t * 32 + q * 16 + lr] = (__bf16)p;    // accumulator layout -> LDS (t rows)
      }
    }
    asm volatile("s_wait_dscnt 0" ::: "memory"); // wave-local LDS relayout handoff
    v16bf pf = cat8(*(const v8bf*)(st + lr * 32 + 16 * hi),
                    *(const v8bf*)(st + lr * 32 + 16 * hi + 8));  // A-layout P fragment
#pragma unroll
    for (int j = 0; j < 4; ++j) {
      const __bf16* vp = Vt + (size_t)(h2 * 64 + 16 * j + lr) * 8192 + n0 + 16 * hi;
      v16bf bv = *(const v16bf*)vp;
      accAV[j] = WMMA_BF16(pf, bv, accAV[j]);
    }
  }
#pragma unroll
  for (int j = 0; j < 4; ++j)
#pragma unroll
    for (int v = 0; v < 8; ++v)
      atomicAdd(&outTile[(v + 8 * hi) * 64 + 16 * j + lr], accAV[j][v]);
  __syncthreads();

  for (int i = tid; i < 1024; i += 256) {
    const int t = i >> 6, d = i & 63;
    const int tok = t0 + t;
    float val = outTile[i];
    if (mask[tok] == 0) val = query[(size_t)tok * 1024 + h2 * 64 + d];  // h_new = where(mask, h_hat, ht)
    Hbf[(size_t)tok * 1024 + h2 * 64 + d] = (__bf16)val;
  }
}

extern "C" void kernel_launch(void* const* d_in, const int* in_sizes, int n_in,
                              void* d_out, int out_size, void* d_ws, size_t ws_size,
                              hipStream_t stream) {
  (void)in_sizes; (void)n_in; (void)out_size; (void)ws_size;
  const float* query = (const float*)d_in[0];
  const int*   mask  = (const int*)d_in[1];
  const float* E     = (const float*)d_in[2];
  const float* wq_w  = (const float*)d_in[3];
  const float* wq_b  = (const float*)d_in[4];
  const float* wk_w  = (const float*)d_in[5];
  const float* wk_b  = (const float*)d_in[6];
  const float* wv_w  = (const float*)d_in[7];
  const float* wv_b  = (const float*)d_in[8];
  const float* wu_w  = (const float*)d_in[9];
  const float* wu_b  = (const float*)d_in[10];
  float* out = (float*)d_out;

  char* ws = (char*)d_ws;                       // ~78 MB used
  __bf16* Ebf  = (__bf16*)(ws + 0);             // 16 MB  E bf16
  __bf16* Htbf = (__bf16*)(ws + 16777216);      // 2 MB   tokens bf16
  __bf16* wqbf = (__bf16*)(ws + 18874368);      // 2 MB
  __bf16* wkbf = (__bf16*)(ws + 20971520);      // 2 MB
  __bf16* wvbf = (__bf16*)(ws + 23068672);      // 2 MB
  __bf16* wubf = (__bf16*)(ws + 25165824);      // 2 MB
  __bf16* Qbf  = (__bf16*)(ws + 27262976);      // 2 MB   Q projection
  __bf16* Kbf  = (__bf16*)(ws + 29360128);      // 16 MB  K projection
  __bf16* Vbf  = (__bf16*)(ws + 46137344);      // 16 MB  V projection
  __bf16* Vt   = (__bf16*)(ws + 62914560);      // 16 MB  V transposed per head
  __bf16* Hbf  = (__bf16*)(ws + 79691776);      // 2 MB   h_new bf16

  k_cvt<<<8388608 / 256, 256, 0, stream>>>(E,     Ebf,  8388608);
  k_cvt<<<1048576 / 256, 256, 0, stream>>>(query, Htbf, 1048576);
  k_cvt<<<1048576 / 256, 256, 0, stream>>>(wq_w,  wqbf, 1048576);
  k_cvt<<<1048576 / 256, 256, 0, stream>>>(wk_w,  wkbf, 1048576);
  k_cvt<<<1048576 / 256, 256, 0, stream>>>(wv_w,  wvbf, 1048576);
  k_cvt<<<1048576 / 256, 256, 0, stream>>>(wu_w,  wubf, 1048576);

  // projections (bf16 WMMA, f32 accumulate, bf16 out); wave = 64x64 output tile
  k_gemm_bf16<<<32,  256, 0, stream>>>(Htbf, wqbf, wq_b, nullptr, nullptr, Qbf, 1024, 1024, 1024);
  k_gemm_bf16<<<256, 256, 0, stream>>>(Ebf,  wkbf, wk_b, nullptr, nullptr, Kbf, 8192, 1024, 1024);
  k_gemm_bf16<<<256, 256, 0, stream>>>(Ebf,  wvbf, wv_b, nullptr, nullptr, Vbf, 8192, 1024, 1024);
  k_vtrans<<<2048, 256, 0, stream>>>(Vbf, Vt);

  // fused attention-like middle (score never materialized; 512 MB of HBM traffic avoided)
  dim3 ag(16, 64);
  k_attn<<<ag, 256, 0, stream>>>(Qbf, Kbf, Vt, query, mask, Hbf);

  // out = query + h_new @ wu^T + wu_b  (f32 out with residual)
  k_gemm_bf16<<<32, 256, 0, stream>>>(Hbf, wubf, wu_b, query, out, nullptr, 1024, 1024, 1024);
}
